// QuantumNeuralParameterEstimator_84301618086516
// MI455X (gfx1250) — matrix operand order repaired and checked by reading, not compiled, not run on hardware
//
#include <hip/hip_runtime.h>
#include <math.h>

typedef __attribute__((ext_vector_type(16))) _Float16 v16h;
typedef __attribute__((ext_vector_type(8)))  float    v8f;

#define NQ  12
#define DIM 4096   // 2^12 amplitudes

// ---------------------------------------------------------------------------
// Kernel 1: simulate the fixed (weights-only) part of the circuit applied to
// each of the 8 basis states |b0 b1 b2 0...0>. One workgroup per column;
// the 4096-amp complex state lives entirely in LDS (32 KB).
// Writes M[k][col] as interleaved (re,im) to scratch: M is [4096][8][2] f32.
// ---------------------------------------------------------------------------
__global__ __launch_bounds__(256)
void qnpe_sim_columns(const float* __restrict__ W, float* __restrict__ M)
{
    __shared__ float sr[DIM];
    __shared__ float si[DIM];
    const int tid = threadIdx.x;
    const int col = blockIdx.x;            // 0..7  (b0b1b2)

    for (int k = tid; k < DIM; k += 256) { sr[k] = 0.f; si[k] = 0.f; }
    __syncthreads();
    if (tid == 0) sr[col * 512] = 1.0f;    // index = b0*2048+b1*1024+b2*512
    __syncthreads();

    auto ry = [&](int q, float th) {
        float c = cosf(0.5f * th), s = sinf(0.5f * th);
        const int sh = 11 - q, stride = 1 << sh;
        for (int p = tid; p < DIM / 2; p += 256) {
            int i0 = ((p >> sh) << (sh + 1)) | (p & (stride - 1));
            int i1 = i0 | stride;
            float r0 = sr[i0], r1 = sr[i1], m0 = si[i0], m1 = si[i1];
            sr[i0] = c * r0 - s * r1;  sr[i1] = s * r0 + c * r1;
            si[i0] = c * m0 - s * m1;  si[i1] = s * m0 + c * m1;
        }
        __syncthreads();
    };
    auto rz = [&](int q, float th) {
        float c = cosf(0.5f * th), s = sinf(0.5f * th);
        const int sh = 11 - q;
        for (int i = tid; i < DIM; i += 256) {
            float sg = ((i >> sh) & 1) ? 1.f : -1.f;   // bit0: e^{-i th/2}
            float r = sr[i], m = si[i];
            sr[i] = c * r - sg * s * m;
            si[i] = c * m + sg * s * r;
        }
        __syncthreads();
    };
    auto cnot = [&](int cq, int tq) {
        const int bc = 11 - cq, bt = 11 - tq;
        const int lo = bc < bt ? bc : bt, hi = bc < bt ? bt : bc;
        const int midbits = hi - 1 - lo;
        const int tmask = 1 << bt, cmask = 1 << bc;
        for (int p = tid; p < DIM / 4; p += 256) {
            int pl  = p & ((1 << lo) - 1);
            int pm  = (p >> lo) & ((1 << midbits) - 1);
            int phh = p >> (lo + midbits);
            int i0 = pl | (pm << (lo + 1)) | (phh << (hi + 1)) | cmask; // bit bt = 0
            int i1 = i0 | tmask;
            float tr = sr[i0]; sr[i0] = sr[i1]; sr[i1] = tr;
            float ti = si[i0]; si[i0] = si[i1]; si[i1] = ti;
        }
        __syncthreads();
    };
    auto rot_layer = [&](int ryRow, int rzRow) {
        for (int q = 0; q < NQ; ++q) {
            ry(q, W[ryRow * 12 + q]);
            if (rzRow >= 0) rz(q, W[rzRow * 12 + q]);
        }
    };

    rot_layer(0, 1);
    for (int c = 0; c < 3; ++c) for (int t = 3; t < 8;  ++t) cnot(c, t);
    for (int c = 0; c < 3; ++c) for (int t = 8; t < 12; ++t) cnot(c, t);
    rot_layer(2, 3);
    {
        const int pk[5] = {3, 4, 5, 6, 7};
        for (int i = 0; i < 5; ++i) cnot(pk[i], pk[(i + 1) % 5]);
        const int pd[4] = {8, 9, 10, 11};
        for (int i = 0; i < 4; ++i) cnot(pd[i], pd[(i + 1) % 4]);
    }
    rot_layer(4, 5);
    {
        const int cs[5] = {3, 4, 5, 6, 7};
        const int ts[5] = {8, 9, 10, 11, 8};
        for (int i = 0; i < 5; ++i) cnot(cs[i], ts[i]);
    }
    rot_layer(6, 7);
    rot_layer(8, -1);

    for (int k = tid; k < DIM; k += 256) {
        M[(k * 8 + col) * 2 + 0] = sr[k];
        M[(k * 8 + col) * 2 + 1] = si[k];
    }
}

// ---------------------------------------------------------------------------
// Kernel 2: Gram matrices Q_q = Re(M^H S_q M), q = 0..8 (qubits 3..11).
// One block per q. Qflat stored as [64][9] f32 (k = j*8+j').
// ---------------------------------------------------------------------------
__global__ __launch_bounds__(256)
void qnpe_build_Q(const float* __restrict__ M, float* __restrict__ Q)
{
    const int q = blockIdx.x;        // 0..8
    const int tid = threadIdx.x;
    const int sh = 8 - q;            // sign bit = (s >> (11-(q+3))) & 1

    float acc[64];
#pragma unroll
    for (int k = 0; k < 64; ++k) acc[k] = 0.f;

    for (int s = tid; s < DIM; s += 256) {
        float sg = ((s >> sh) & 1) ? -1.f : 1.f;
        const float* row = M + s * 16;
        float re[8], im[8];
#pragma unroll
        for (int j = 0; j < 8; ++j) { re[j] = row[2 * j]; im[j] = row[2 * j + 1]; }
#pragma unroll
        for (int j = 0; j < 8; ++j)
#pragma unroll
            for (int j2 = 0; j2 < 8; ++j2)
                acc[j * 8 + j2] += sg * (re[j] * re[j2] + im[j] * im[j2]);
    }

    __shared__ float red[64];
    if (tid < 64) red[tid] = 0.f;
    __syncthreads();
#pragma unroll
    for (int k = 0; k < 64; ++k) atomicAdd(&red[k], acc[k]);   // ds_add_f32
    __syncthreads();
    if (tid < 64) Q[tid * 9 + q] = red[tid];
}

// ---------------------------------------------------------------------------
// Kernel 3: per-sample evaluation via WMMA.
//   a8 from the 3 input angles; z = a (x) a (64 f16); E = Z @ Qflat via two
//   chained v_wmma_f32_16x16x32_f16 (K = 64 split 32+32, N = 16, 9 used),
//   then affine bounds mapping. One wave per 16-sample tile, 8 waves/block.
// ---------------------------------------------------------------------------
__global__ __launch_bounds__(256)
void qnpe_eval(const float* __restrict__ cov, const float* __restrict__ dose,
               const float* __restrict__ Q,   const float* __restrict__ pkB,
               const float* __restrict__ pdB, float* __restrict__ out, int B)
{
    __shared__ _Float16 qlds[64][16];          // Qflat, zero-padded to N=16
    __shared__ _Float16 zlds[8][16][64];       // per-wave Z tiles
    const int tid = threadIdx.x;

    for (int idx = tid; idx < 64 * 16; idx += 256) {
        int k = idx >> 4, c = idx & 15;
        qlds[k][c] = (c < 9) ? (_Float16)Q[k * 9 + c] : (_Float16)0.f;
    }

    const int wave = tid >> 5, lane = tid & 31;
    const int tileBase = blockIdx.x * 128 + wave * 16;

    if (lane < 16) {
        const int i = tileBase + lane;
        float a[8];
        if (i < B) {
            float x0 = cov[2 * i], x1 = cov[2 * i + 1], x2 = dose[i];
            float c0 = cosf(0.5f * x0), s0 = sinf(0.5f * x0);
            float c1 = cosf(0.5f * x1), s1 = sinf(0.5f * x1);
            float c2 = cosf(0.5f * x2), s2 = sinf(0.5f * x2);
#pragma unroll
            for (int j = 0; j < 8; ++j) {
                float f0 = (j & 4) ? s0 : c0;
                float f1 = (j & 2) ? s1 : c1;
                float f2 = (j & 1) ? s2 : c2;
                a[j] = f0 * f1 * f2;
            }
        } else {
#pragma unroll
            for (int j = 0; j < 8; ++j) a[j] = 0.f;
        }
#pragma unroll
        for (int j = 0; j < 8; ++j)
#pragma unroll
            for (int j2 = 0; j2 < 8; ++j2)
                zlds[wave][lane][j * 8 + j2] = (_Float16)(a[j] * a[j2]);
    }
    __syncthreads();

    const int m = lane & 15, hi = lane >> 4, n = m;
    v16h aop0, aop1, bop0, bop1;
#pragma unroll
    for (int e = 0; e < 16; ++e) {
        // A in this code's contiguous k order (same as B): lanes 0-15 -> K 0..15; lanes 16-31 -> K 16..31
        int kA = e + hi * 16;
        aop0[e] = zlds[wave][m][kA];
        aop1[e] = zlds[wave][m][kA + 32];
        // B in this code's contiguous k order (same as A): lanes 0-15 -> K 0..15; lanes 16-31 -> K 16..31
        int kB = e + hi * 16;
        bop0[e] = qlds[kB][n];
        bop1[e] = qlds[kB + 32][n];
    }

    v8f acc = {};
    acc = __builtin_amdgcn_wmma_f32_16x16x32_f16(false, aop0, false, bop0,
                                                 (short)0, acc, false, false);
    acc = __builtin_amdgcn_wmma_f32_16x16x32_f16(false, aop1, false, bop1,
                                                 (short)0, acc, false, false);

    // D layout: N = lane%16 ; VGPR r -> M = r + 8*(lane>=16)
    if (n < 9) {
        float lo, hiB;
        if (n < 5) { lo = pkB[2 * n];       hiB = pkB[2 * n + 1]; }
        else       { lo = pdB[2 * (n - 5)]; hiB = pdB[2 * (n - 5) + 1]; }
#pragma unroll
        for (int r = 0; r < 8; ++r) {
            int i = tileBase + r + hi * 8;
            if (i < B) {
                float ev  = acc[r];
                float val = lo + (ev + 1.f) * 0.5f * (hiB - lo);
                if (n < 5) out[i * 5 + n]               = val;   // pk (B,5)
                else       out[B * 5 + i * 4 + (n - 5)] = val;   // pd (B,4)
            }
        }
    }
}

// ---------------------------------------------------------------------------
extern "C" void kernel_launch(void* const* d_in, const int* in_sizes, int n_in,
                              void* d_out, int out_size, void* d_ws, size_t ws_size,
                              hipStream_t stream)
{
    // inputs: subject_ids(B) [unused], covariates(B,2), dose(B), weights(108),
    //         pk_bounds(5,2), pd_bounds(4,2)
    const float* cov  = (const float*)d_in[1];
    const float* dose = (const float*)d_in[2];
    const float* W    = (const float*)d_in[3];
    const float* pkB  = (const float*)d_in[4];
    const float* pdB  = (const float*)d_in[5];
    float* out = (float*)d_out;
    const int B = in_sizes[0];

    float* M = (float*)d_ws;        // [4096][8][2] f32 = 256 KB
    float* Q = M + DIM * 16;        // [64][9] f32

    qnpe_sim_columns<<<8, 256, 0, stream>>>(W, M);
    qnpe_build_Q   <<<9, 256, 0, stream>>>(M, Q);
    const int blocks = (B + 127) / 128;
    qnpe_eval<<<blocks, 256, 0, stream>>>(cov, dose, Q, pkB, pdB, out, B);
}
